// WindowAttention_43130061586462
// MI455X (gfx1250) — matrix operand order repaired and checked by reading, not compiled, not run on hardware
//
#include <hip/hip_runtime.h>
#include <hip/hip_bf16.h>

typedef __bf16 bf16_t;
typedef __attribute__((ext_vector_type(16))) __bf16 v16bf;
typedef __attribute__((ext_vector_type(8)))  __bf16 v8bf;
typedef __attribute__((ext_vector_type(4)))  __bf16 v4bf;
typedef __attribute__((ext_vector_type(8)))  float  v8f;

#define DIM      512
#define NTOK     64
#define HEADS    16
#define HEAD_DIM 32
#define NWIN     2048
#define ATT_SCALE 0.17677669529663687f  // 32^-0.5

// ---------------------------------------------------------------------------
// WMMA fragment helpers, per CDNA5 ISA 16-bit layouts (CDNA5 ISA):
//  A 16x32: lane L(0-15) row M=L holds K {0..7,16..23}; lane L+16 holds {8..15,24..31}
//  B 32x16: lane L(0-15) col N=L holds K {0..7,16..23}; lane L+16 holds {8..15,24..31}
//  C/D 16x16 f32: lane = (M/8)*16 + N, vgpr r = M%8
// ---------------------------------------------------------------------------
__device__ __forceinline__ v16bf load_a_frag(const bf16_t* A, int lda, int lane) {
    int l = lane & 15, hi = lane >> 4;
    const bf16_t* p = A + l * lda + hi * 8;
    v16bf r;
    ((v8bf*)&r)[0] = *(const v8bf*)(p);
    ((v8bf*)&r)[1] = *(const v8bf*)(p + 16);
    return r;
}

__device__ __forceinline__ v16bf load_b_frag(const bf16_t* W, int ldw, int lane) {
    int l = lane & 15, hi = lane >> 4;
    const bf16_t* p = W + l * ldw + hi * 8;
    v16bf r;
    ((v8bf*)&r)[0] = *(const v8bf*)(p);
    ((v8bf*)&r)[1] = *(const v8bf*)(p + 16);
    return r;
}

__device__ __forceinline__ v8f wmma_bf16(v16bf a, v16bf b, v8f c) {
    return __builtin_amdgcn_wmma_f32_16x16x32_bf16(
        false, a, false, b, (short)0, c, false, false);
}

// CDNA5 async copy global->LDS (16B per lane, ASYNCcnt-tracked)
__device__ __forceinline__ void async_copy_b128(unsigned lds_byte_addr,
                                                const void* gaddr) {
    asm volatile("global_load_async_to_lds_b128 %0, %1, off"
                 :: "v"(lds_byte_addr), "v"(gaddr) : "memory");
}
__device__ __forceinline__ void wait_async0() {
    asm volatile("s_wait_asynccnt 0x0" ::: "memory");
}

// ---------------------------------------------------------------------------
// Prep: weights fp32 -> bf16 (fused [q_w ; kv_w] = 1536x512, proj_w = 512x512)
// ---------------------------------------------------------------------------
__global__ void prep_weights(const float* __restrict__ q_w,
                             const float* __restrict__ kv_w,
                             const float* __restrict__ proj_w,
                             bf16_t* __restrict__ w1, bf16_t* __restrict__ w2) {
    int idx = blockIdx.x * blockDim.x + threadIdx.x;
    const int W1E = 1536 * DIM;
    if (idx < W1E) {
        int row = idx >> 9, k = idx & 511;
        float v = (row < DIM) ? q_w[row * DIM + k] : kv_w[(row - DIM) * DIM + k];
        w1[idx] = (bf16_t)v;
    } else {
        int j = idx - W1E;
        if (j < DIM * DIM) w2[j] = (bf16_t)proj_w[j];
    }
}

// Relative-position bias gather -> bias[h][i][j], fp32 (16x64x64)
__global__ void prep_bias(const float* __restrict__ bias_table,
                          const int* __restrict__ rel_index,
                          float* __restrict__ bias) {
    int idx = blockIdx.x * blockDim.x + threadIdx.x;  // 0..4095 = i*64+j
    if (idx >= NTOK * NTOK) return;
    int r = rel_index[idx];
    #pragma unroll
    for (int h = 0; h < HEADS; ++h)
        bias[h * NTOK * NTOK + idx] = bias_table[r * HEADS + h];
}

// ---------------------------------------------------------------------------
// Fused QKV projection: per-window GEMM  C[64x1536] = q[64x512] @ W1^T
// One block per window, 8 waves; A tile converted fp32->bf16 into LDS.
// Outputs qh/k/v stored (B,N,H,HD) with HD innermost.
// ---------------------------------------------------------------------------
__global__ __launch_bounds__(256) void qkv_gemm(
    const float* __restrict__ q, const bf16_t* __restrict__ w1,
    const float* __restrict__ q_b, const float* __restrict__ kv_b,
    bf16_t* __restrict__ qh, bf16_t* __restrict__ kk, bf16_t* __restrict__ vv) {
    __shared__ bf16_t sA[NTOK][DIM];  // 64 KB
    int b = blockIdx.x, tid = threadIdx.x;
    const float4* q4 = (const float4*)(q + (size_t)b * NTOK * DIM);
    for (int i = tid; i < NTOK * DIM / 4; i += 256) {
        float4 f = q4[i];
        v4bf o = {(bf16_t)f.x, (bf16_t)f.y, (bf16_t)f.z, (bf16_t)f.w};
        *(v4bf*)(&sA[0][0] + i * 4) = o;   // 8B ds_store
    }
    __syncthreads();

    int w = tid >> 5, lane = tid & 31;
    int l = lane & 15, hi = lane >> 4;

    for (int np = 0; np < 6; ++np) {        // 2 n-tiles per pass, 12 total/wave
        v8f acc[2][4];
        v8f z = {};
        #pragma unroll
        for (int j = 0; j < 2; ++j)
            #pragma unroll
            for (int mt = 0; mt < 4; ++mt) acc[j][mt] = z;
        int n0 = w * 192 + np * 32;
        if (np < 5)  // hint the next pass' weight rows toward the caches
            __builtin_prefetch(w1 + (size_t)(n0 + 32) * DIM, 0, 1);
        for (int kt = 0; kt < 16; ++kt) {
            int k0 = kt * 32;
            v16bf af[4];
            #pragma unroll
            for (int mt = 0; mt < 4; ++mt)
                af[mt] = load_a_frag(&sA[mt * 16][k0], DIM, lane);
            #pragma unroll
            for (int j = 0; j < 2; ++j) {
                v16bf bfr = load_b_frag(w1 + (size_t)(n0 + j * 16) * DIM + k0, DIM, lane);
                #pragma unroll
                for (int mt = 0; mt < 4; ++mt)
                    acc[j][mt] = wmma_bf16(af[mt], bfr, acc[j][mt]);
            }
        }
        #pragma unroll
        for (int j = 0; j < 2; ++j) {
            int ntile = n0 + j * 16;           // wave-uniform tile base;
            bf16_t* dst; int nbase; const float* bv;
            if (ntile < DIM)          { dst = qh; nbase = ntile;           bv = q_b + ntile; }
            else if (ntile < 2 * DIM) { dst = kk; nbase = ntile - DIM;     bv = kv_b + (ntile - DIM); }
            else                      { dst = vv; nbase = ntile - 2 * DIM; bv = kv_b + (ntile - DIM); }
            float bn = bv[l];
            #pragma unroll
            for (int mt = 0; mt < 4; ++mt)
                #pragma unroll
                for (int r = 0; r < 8; ++r) {
                    int m = mt * 16 + r + hi * 8;
                    dst[((size_t)b * NTOK + m) * DIM + nbase + l] =
                        (bf16_t)(acc[j][mt][r] + bn);
                }
        }
    }
}

// ---------------------------------------------------------------------------
// Attention: one block per (window, head), 4 waves (128 threads).
// S = scale*qh@k^T + bias + mask ; softmax ; attn out ; O = P@v -> x (bf16)
// ---------------------------------------------------------------------------
#define LDP 72   // padded leading dim (rows stay 16B aligned: 72*2=144)
__global__ __launch_bounds__(128) void attn_kernel(
    const bf16_t* __restrict__ qh, const bf16_t* __restrict__ kk,
    const bf16_t* __restrict__ vv, const float* __restrict__ bias,
    const float* __restrict__ mask, float* __restrict__ attn_out,
    bf16_t* __restrict__ x) {
    __shared__ float  sS[NTOK][NTOK];        // 16 KB logits / exp
    __shared__ bf16_t sP[NTOK][LDP];         // probabilities (bf16), padded
    __shared__ bf16_t sVT[HEAD_DIM][LDP];    // v transposed [hd][token], padded
    __shared__ float  sInv[NTOK];
    int b = blockIdx.x >> 4;
    int h = blockIdx.x & 15;
    int tid = threadIdx.x;

    // stage v^T: consecutive lanes read consecutive hd -> 64B coalesced
    for (int idx = tid; idx < HEAD_DIM * NTOK; idx += 128) {
        int j  = idx >> 5;        // token
        int hd = idx & 31;
        sVT[hd][j] = vv[((size_t)b * NTOK + j) * DIM + h * HEAD_DIM + hd];
    }

    int w = tid >> 5, lane = tid & 31;
    int l = lane & 15, hi = lane >> 4;

    // S slab: rows [w*16, w*16+16), K = HEAD_DIM = 32 -> single WMMA per tile
    v16bf aq = load_a_frag(qh + ((size_t)b * NTOK + w * 16) * DIM + h * HEAD_DIM,
                           DIM, lane);
    const float* biash = bias + h * NTOK * NTOK;
    const float* maskw = mask + (size_t)(b & 63) * NTOK * NTOK;
    for (int ct = 0; ct < 4; ++ct) {
        v16bf bk = load_b_frag(kk + ((size_t)b * NTOK + ct * 16) * DIM + h * HEAD_DIM,
                               DIM, lane);
        v8f acc = {};
        acc = wmma_bf16(aq, bk, acc);
        int jc = ct * 16 + l;
        #pragma unroll
        for (int r = 0; r < 8; ++r) {
            int i = w * 16 + r + hi * 8;
            sS[i][jc] = acc[r] * ATT_SCALE + biash[i * NTOK + jc] + maskw[i * NTOK + jc];
        }
    }
    __syncthreads();

    // fp32 softmax stats: one thread per row (exp left in sS, 1/sum in sInv)
    if (tid < NTOK) {
        int i = tid;
        float mx = -3.0e38f;
        for (int j = 0; j < NTOK; ++j) mx = fmaxf(mx, sS[i][j]);
        float sum = 0.f;
        for (int j = 0; j < NTOK; ++j) {
            float e = __expf(sS[i][j] - mx);
            sS[i][j] = e; sum += e;
        }
        sInv[i] = 1.f / sum;
    }
    __syncthreads();

    // cooperative normalize: coalesced float4 stores of attn + bf16 P fill
    float* ao = attn_out + ((size_t)b * HEADS + h) * NTOK * NTOK;
    for (int idx = tid * 4; idx < NTOK * NTOK; idx += 128 * 4) {
        int i = idx >> 6, j = idx & 63;
        float inv = sInv[i];
        float4 pv;
        pv.x = sS[i][j + 0] * inv;
        pv.y = sS[i][j + 1] * inv;
        pv.z = sS[i][j + 2] * inv;
        pv.w = sS[i][j + 3] * inv;
        v4bf pb = {(bf16_t)pv.x, (bf16_t)pv.y, (bf16_t)pv.z, (bf16_t)pv.w};
        *(v4bf*)(&sP[i][j]) = pb;
        *(float4*)(ao + idx) = pv;
    }
    __syncthreads();

    // O = P @ v : 16x32 slab per wave, K=64 -> 2 WMMA per 16x16 tile
    v16bf a0 = load_a_frag(&sP[w * 16][0],  LDP, lane);
    v16bf a1 = load_a_frag(&sP[w * 16][32], LDP, lane);
    #pragma unroll
    for (int nt = 0; nt < 2; ++nt) {
        v8f acc = {};
        v16bf b0 = load_b_frag(&sVT[nt * 16][0],  LDP, lane);
        acc = wmma_bf16(a0, b0, acc);
        v16bf b1 = load_b_frag(&sVT[nt * 16][32], LDP, lane);
        acc = wmma_bf16(a1, b1, acc);
        #pragma unroll
        for (int r = 0; r < 8; ++r) {
            int m = w * 16 + r + hi * 8;
            x[((size_t)b * NTOK + m) * DIM + h * HEAD_DIM + nt * 16 + l] = (bf16_t)acc[r];
        }
    }
}

// ---------------------------------------------------------------------------
// Output projection: per-window GEMM  out[64x512] = x[64x512] @ proj_w^T + b
// A tile staged via CDNA5 async global->LDS copy (ASYNCcnt path).
// ---------------------------------------------------------------------------
__global__ __launch_bounds__(256) void proj_gemm(
    const bf16_t* __restrict__ x, const bf16_t* __restrict__ w2,
    const float* __restrict__ proj_b, float* __restrict__ out) {
    __shared__ bf16_t sA[NTOK][DIM];  // 64 KB
    int b = blockIdx.x, tid = threadIdx.x;
    {
        const char* src = (const char*)(x + (size_t)b * NTOK * DIM);
        unsigned lds_base = (unsigned)(size_t)(&sA[0][0]);
        for (int i = tid; i < NTOK * DIM * 2 / 16; i += 256)
            async_copy_b128(lds_base + i * 16, src + i * 16);
        wait_async0();
    }
    __syncthreads();

    int w = tid >> 5, lane = tid & 31;
    int l = lane & 15, hi = lane >> 4;

    for (int np = 0; np < 2; ++np) {
        v8f acc[2][4];
        v8f z = {};
        #pragma unroll
        for (int j = 0; j < 2; ++j)
            #pragma unroll
            for (int mt = 0; mt < 4; ++mt) acc[j][mt] = z;
        int n0 = w * 64 + np * 32;
        if (np == 0)
            __builtin_prefetch(w2 + (size_t)(n0 + 32) * DIM, 0, 1);
        for (int kt = 0; kt < 16; ++kt) {
            int k0 = kt * 32;
            v16bf af[4];
            #pragma unroll
            for (int mt = 0; mt < 4; ++mt)
                af[mt] = load_a_frag(&sA[mt * 16][k0], DIM, lane);
            #pragma unroll
            for (int j = 0; j < 2; ++j) {
                v16bf bfr = load_b_frag(w2 + (size_t)(n0 + j * 16) * DIM + k0, DIM, lane);
                #pragma unroll
                for (int mt = 0; mt < 4; ++mt)
                    acc[j][mt] = wmma_bf16(af[mt], bfr, acc[j][mt]);
            }
        }
        #pragma unroll
        for (int j = 0; j < 2; ++j) {
            int n = n0 + j * 16 + l;
            float bn = proj_b[n];
            #pragma unroll
            for (int mt = 0; mt < 4; ++mt)
                #pragma unroll
                for (int r = 0; r < 8; ++r) {
                    int m = mt * 16 + r + hi * 8;
                    out[((size_t)b * NTOK + m) * DIM + n] = acc[j][mt][r] + bn;
                }
        }
    }
}

extern "C" void kernel_launch(void* const* d_in, const int* in_sizes, int n_in,
                              void* d_out, int out_size, void* d_ws, size_t ws_size,
                              hipStream_t stream) {
    (void)in_sizes; (void)n_in; (void)out_size; (void)ws_size;
    const float* q          = (const float*)d_in[0];
    const float* mask       = (const float*)d_in[1];
    const float* q_w        = (const float*)d_in[2];
    const float* q_b        = (const float*)d_in[3];
    const float* kv_w       = (const float*)d_in[4];
    const float* kv_b       = (const float*)d_in[5];
    const float* proj_w     = (const float*)d_in[6];
    const float* proj_b     = (const float*)d_in[7];
    const float* bias_table = (const float*)d_in[8];
    const int*   rel_index  = (const int*)d_in[9];

    // Workspace layout (~540 MB total):
    char* ws = (char*)d_ws;
    bf16_t* w1    = (bf16_t*)(ws);                 // 1536*512*2 = 1.5 MB
    bf16_t* w2    = (bf16_t*)(ws + 1572864);       // 512*512*2  = 0.5 MB
    float*  biasb = (float*) (ws + 2097152);       // 16*64*64*4 = 256 KB
    size_t  big   = (size_t)NWIN * NTOK * DIM * sizeof(bf16_t);  // 134 MB each
    bf16_t* qh    = (bf16_t*)(ws + 2359296);
    bf16_t* kk    = (bf16_t*)((char*)qh + big);
    bf16_t* vv    = (bf16_t*)((char*)kk + big);
    bf16_t* x     = (bf16_t*)((char*)vv + big);

    float* out_x    = (float*)d_out;                               // (B,N,DIM)
    float* out_attn = out_x + (size_t)NWIN * NTOK * DIM;           // (B,H,N,N)

    prep_weights<<<(1536 * DIM + DIM * DIM) / 256, 256, 0, stream>>>(
        q_w, kv_w, proj_w, w1, w2);
    prep_bias<<<(NTOK * NTOK + 255) / 256, 256, 0, stream>>>(
        bias_table, rel_index, biasb);
    qkv_gemm<<<NWIN, 256, 0, stream>>>(q, w1, q_b, kv_b, qh, kk, vv);
    attn_kernel<<<NWIN * HEADS, 128, 0, stream>>>(
        qh, kk, vv, biasb, mask, out_attn, x);
    proj_gemm<<<NWIN, 256, 0, stream>>>(x, w2, proj_b, out_x);
}
